// GraphLayer_62646392980172
// MI455X (gfx1250) — matrix operand order repaired and checked by reading, not compiled, not run on hardware
//
#include <hip/hip_runtime.h>
#include <hip/hip_bf16.h>

// CDNA5 / gfx1250: wave32, WMMA 16x16x32 bf16, vectorized b128 fragment moves.
typedef __attribute__((ext_vector_type(16))) __bf16 v16bf;
typedef __attribute__((ext_vector_type(8)))  float  v8f;

#define DEV static __device__ __forceinline__

DEV __bf16 f2bf(float f) {
  unsigned u = __float_as_uint(f);
  unsigned r = (u + 0x7FFFu + ((u >> 16) & 1u)) >> 16;   // round-to-nearest-even
  unsigned short s = (unsigned short)r;
  return __builtin_bit_cast(__bf16, s);
}

struct bfx16 { uint4 lo, hi; };
DEV v16bf cat16(uint4 lo, uint4 hi) { bfx16 t{lo, hi}; return __builtin_bit_cast(v16bf, t); }

DEV ushort4 pack4(float a, float b, float c, float d) {
  ushort4 w;
  w.x = __builtin_bit_cast(unsigned short, f2bf(a));
  w.y = __builtin_bit_cast(unsigned short, f2bf(b));
  w.z = __builtin_bit_cast(unsigned short, f2bf(c));
  w.w = __builtin_bit_cast(unsigned short, f2bf(d));
  return w;
}

DEV float eluf(float v) { return v > 0.f ? v : (__expf(v) - 1.f); }

DEV void atomic_add_f32(float* p, float v) {
  __hip_atomic_fetch_add(p, v, __ATOMIC_RELAXED, __HIP_MEMORY_SCOPE_AGENT);
}

// ---------------------------------------------------------------------------
// W1 prep: f32 [d_cat,16] -> bf16 in WMMA B-fragment order:
//   dst[((s*32 + lane)*16) + t] = W1[ksrc(kpad)][lane&15],
//   kpad = s*32 + (lane>>4)*8 + (t&7) + (t>>3)*16.
// mode 0 (node): ksrc = kpad < d ? kpad : zero  (identity layout)
// mode 1 (edge): concat layout [0..128]=efeat, [132..259]=n1, [260..387]=n2
__global__ void prep_w1_frag(const float* __restrict__ src, __bf16* __restrict__ dst,
                             int kp, int mode, int d) {
  int i = blockIdx.x * blockDim.x + threadIdx.x;
  if (i >= kp * 16) return;
  int t = i & 15, lane = (i >> 4) & 31, s = i >> 9;
  int kpad = s * 32 + ((lane >> 4) << 3) + (t & 7) + ((t >> 3) << 4);
  int n = lane & 15;
  int ks;
  if (mode == 0) ks = (kpad < d) ? kpad : -1;
  else           ks = (kpad < 129) ? kpad : (kpad >= 132 && kpad < 388) ? kpad - 3 : -1;
  dst[i] = f2bf(ks >= 0 ? src[ks * 16 + n] : 0.f);
}

// W2 prep: f32 [16,D2] -> bf16 B-fragments, Dt tiles of 16 cols, K padded 16->32.
__global__ void prep_w2_frag(const float* __restrict__ src, __bf16* __restrict__ dst,
                             int D2, int Dt) {
  int i = blockIdx.x * blockDim.x + threadIdx.x;
  if (i >= Dt * 512) return;
  int t = i & 15, lane = (i >> 4) & 31, j = i >> 9;
  int K = ((lane >> 4) << 3) + (t & 7) + ((t >> 3) << 4);
  int c = j * 16 + (lane & 15);
  dst[i] = f2bf((K < 16 && c < D2) ? src[K * D2 + c] : 0.f);
}

__global__ void zero_kernel(float* __restrict__ p, long long n) {
  long long i = (long long)blockIdx.x * blockDim.x + threadIdx.x;
  if (i < n) p[i] = 0.f;
}

// ---------------------------------------------------------------------------
// Edge MLPs: cat(LDS layout) = [efeat 0..128 | 0 0 0 | n1 132..259 | n2 260..387 | 0...]
#define EKP 416
#define ESTEPS 13

__global__ __launch_bounds__(128)
void edge_mlp_kernel(const float* __restrict__ efeat,   // [E,129]
                     const float* __restrict__ nfeat,   // [N,128] (16B-aligned rows)
                     const long long* __restrict__ gi1,
                     const long long* __restrict__ gi2,
                     const __bf16* __restrict__ w1mf, const float* __restrict__ b1m,
                     const __bf16* __restrict__ w2mf, const float* __restrict__ b2m,
                     int D2, int Dt,
                     const __bf16* __restrict__ w1af, const float* __restrict__ b1a,
                     const float* __restrict__ w2a,  const float* __restrict__ b2a,
                     float* __restrict__ msg_out, float* __restrict__ att_out, int E) {
  __shared__ __align__(16) __bf16 catl[4][16 * EKP];
  __shared__ __align__(16) __bf16 hm[4][256];
  __shared__ __align__(16) __bf16 ha[4][256];

  const int wave  = threadIdx.x >> 5;
  const int lane  = threadIdx.x & 31;
  const int tile0 = (blockIdx.x * 4 + wave) * 16;

  __bf16* my = catl[wave];
  const __bf16 zb = f2bf(0.f);
  for (int r = 0; r < 16; ++r) {
    int e = tile0 + r; if (e >= E) e = 0;
    const float* pe = efeat + (long long)e * 129;   // rows only 4B-aligned -> scalar
    const float* p1 = nfeat + gi1[e] * 128;         // 512B rows -> float4
    const float* p2 = nfeat + gi2[e] * 128;
    __bf16* dst = my + r * EKP;
    for (int k = lane; k < 132; k += 32) dst[k] = (k < 129) ? f2bf(pe[k]) : zb;
    float4 v1 = ((const float4*)p1)[lane];
    *(ushort4*)(dst + 132 + 4 * lane) = pack4(v1.x, v1.y, v1.z, v1.w);
    float4 v2 = ((const float4*)p2)[lane];
    *(ushort4*)(dst + 260 + 4 * lane) = pack4(v2.x, v2.y, v2.z, v2.w);
    if (lane < 28) dst[388 + lane] = zb;
  }
  __syncthreads();

  const int n  = lane & 15;   // D column (hidden unit); also A row M
  const int hk = lane >> 4;   // lane half -> K sub-range
  const __bf16* arow = my + n * EKP + (hk << 3);
  v8f accm = {}; v8f acca = {};
  for (int s = 0; s < ESTEPS; ++s) {
    const uint4 alo = *(const uint4*)(arow + s * 32);        // K = kb+hk*8 .. +7
    const uint4 ahi = *(const uint4*)(arow + s * 32 + 16);   // K = kb+16+hk*8 ..
    const v16bf a  = cat16(alo, ahi);
    const v16bf bm = *(const v16bf*)(w1mf + ((s * 32 + lane) << 4));
    const v16bf ba = *(const v16bf*)(w1af + ((s * 32 + lane) << 4));
    accm = __builtin_amdgcn_wmma_f32_16x16x32_bf16(false, a, false, bm, (short)0, accm, false, false);
    acca = __builtin_amdgcn_wmma_f32_16x16x32_bf16(false, a, false, ba, (short)0, acca, false, false);
  }
  const float bbm = b1m[n], bba = b1a[n];
#pragma unroll
  for (int r = 0; r < 8; ++r) {
    const int mm = r + (hk << 3);
    hm[wave][mm * 16 + n] = f2bf(eluf(accm[r] + bbm));
    ha[wave][mm * 16 + n] = f2bf(eluf(acca[r] + bba));
  }
  __syncthreads();

  // --- second layer (message) via WMMA: A = h (16x32, upper K-half zero) ---
  const uint4 z4 = make_uint4(0u, 0u, 0u, 0u);
  const v16bf a2 = cat16(*(const uint4*)(&hm[wave][n * 16 + (hk << 3)]), z4);
  for (int j = 0; j < Dt; ++j) {
    const v16bf b2v = *(const v16bf*)(w2mf + ((j * 32 + lane) << 4));
    v8f acc2 = {};
    acc2 = __builtin_amdgcn_wmma_f32_16x16x32_bf16(false, a2, false, b2v, (short)0, acc2, false, false);
    const int c = j * 16 + n;
    if (c < D2) {
      const float bias = b2m[c];
#pragma unroll
      for (int r = 0; r < 8; ++r) {
        const int e2 = tile0 + r + (hk << 3);
        if (e2 < E) msg_out[(long long)e2 * D2 + c] = acc2[r] + bias;
      }
    }
  }
  // --- attention head (1-wide): VALU dot-16 ---
  if (hk == 0) {
    const int e = tile0 + lane;
    if (e < E) {
      float sum = b2a[0];
      const __bf16* h1 = &ha[wave][lane * 16];
#pragma unroll
      for (int k = 0; k < 16; ++k) sum += (float)h1[k] * w2a[k];
      att_out[e] = sum;
    }
  }
}

// ---------------------------------------------------------------------------
// Node MLP: cat = [in0(d0=128) | in1(d1)] -> 16(ELU) -> 128 (identity layout)
#define NKPMAX 288

__global__ __launch_bounds__(128)
void node_mlp_kernel(const float* __restrict__ in0,
                     const float* __restrict__ in1, int d1,
                     const __bf16* __restrict__ w1f, const float* __restrict__ b1,
                     const __bf16* __restrict__ w2f, const float* __restrict__ b2,
                     float* __restrict__ out, int N, int kp) {
  __shared__ __align__(16) __bf16 catl[4][16 * NKPMAX];
  __shared__ __align__(16) __bf16 hm[4][256];
  const int wave  = threadIdx.x >> 5;
  const int lane  = threadIdx.x & 31;
  const int tile0 = (blockIdx.x * 4 + wave) * 16;

  __bf16* my = catl[wave];
  const __bf16 zb = f2bf(0.f);
  for (int r = 0; r < 16; ++r) {
    int nd = tile0 + r; if (nd >= N) nd = 0;
    const float* p0 = in0 + (long long)nd * 128;
    const float* p1 = in1 + (long long)nd * d1;
    __bf16* dst = my + r * kp;
    float4 v0 = ((const float4*)p0)[lane];
    *(ushort4*)(dst + 4 * lane) = pack4(v0.x, v0.y, v0.z, v0.w);
    if (d1 == 128) {           // 512B-aligned rows -> float4
      float4 v1 = ((const float4*)p1)[lane];
      *(ushort4*)(dst + 128 + 4 * lane) = pack4(v1.x, v1.y, v1.z, v1.w);
    } else {
      for (int k = lane; k < d1; k += 32) dst[128 + k] = f2bf(p1[k]);
    }
    for (int k = 128 + d1 + lane; k < kp; k += 32) dst[k] = zb;
  }
  __syncthreads();

  const int n = lane & 15, hk = lane >> 4;
  const __bf16* arow = my + n * kp + (hk << 3);
  v8f acc = {};
  const int steps = kp >> 5;
  for (int s = 0; s < steps; ++s) {
    const v16bf a = cat16(*(const uint4*)(arow + s * 32),
                          *(const uint4*)(arow + s * 32 + 16));
    const v16bf b = *(const v16bf*)(w1f + ((s * 32 + lane) << 4));
    acc = __builtin_amdgcn_wmma_f32_16x16x32_bf16(false, a, false, b, (short)0, acc, false, false);
  }
  const float bb = b1[n];
#pragma unroll
  for (int r = 0; r < 8; ++r)
    hm[wave][(r + (hk << 3)) * 16 + n] = f2bf(eluf(acc[r] + bb));
  __syncthreads();

  const uint4 z4 = make_uint4(0u, 0u, 0u, 0u);
  const v16bf a2 = cat16(*(const uint4*)(&hm[wave][n * 16 + (hk << 3)]), z4);
  for (int j = 0; j < 8; ++j) {   // 128 output cols = 8 tiles
    const v16bf b2v = *(const v16bf*)(w2f + ((j * 32 + lane) << 4));
    v8f acc2 = {};
    acc2 = __builtin_amdgcn_wmma_f32_16x16x32_bf16(false, a2, false, b2v, (short)0, acc2, false, false);
    const int c = j * 16 + n;
    const float bias = b2[c];
#pragma unroll
    for (int r = 0; r < 8; ++r) {
      const int nd = tile0 + r + (hk << 3);
      if (nd < N) out[(long long)nd * 128 + c] = acc2[r] + bias;
    }
  }
}

// ---------------------------------------------------------------------------
// Scatter softmax (unsorted segments) via atomics.
DEV unsigned ord_map(float f) {
  unsigned u = __float_as_uint(f);
  return u ^ (unsigned)(((int)u >> 31) | 0x80000000);
}
DEV float ord_unmap(unsigned m) {
  unsigned u = m ^ (unsigned)(((int)(~m) >> 31) | 0x80000000);
  return __uint_as_float(u);
}

__global__ void seg_max_kernel(const float* __restrict__ e,
                               const long long* __restrict__ seg,
                               unsigned* __restrict__ mx, int E) {
  int i = blockIdx.x * blockDim.x + threadIdx.x;
  if (i >= E) return;
  atomicMax(&mx[seg[i]], ord_map(e[i]));
}

__global__ void seg_expsum_kernel(const float* __restrict__ e,
                                  const long long* __restrict__ seg,
                                  const unsigned* __restrict__ mx,
                                  float* __restrict__ ex, float* __restrict__ sum, int E) {
  int i = blockIdx.x * blockDim.x + threadIdx.x;
  if (i >= E) return;
  long long s = seg[i];
  float v = __expf(e[i] - ord_unmap(mx[s]));
  ex[i] = v;
  atomic_add_f32(&sum[s], v);
}

__global__ __launch_bounds__(256)
void seg_agg_kernel(const float* __restrict__ msg, const float* __restrict__ ex,
                    const float* __restrict__ sum, const long long* __restrict__ seg,
                    float* __restrict__ agg, int E, int D) {
  const int wave = threadIdx.x >> 5, lane = threadIdx.x & 31;
  const int e = blockIdx.x * 8 + wave;
  if (e >= E) return;
  const long long s = seg[e];
  const float alpha = ex[e] / sum[s];
  const float* m = msg + (long long)e * D;
  float* a = agg + s * D;
  for (int c = lane; c < D; c += 32) atomic_add_f32(&a[c], m[c] * alpha);
}

// ---------------------------------------------------------------------------
extern "C" void kernel_launch(void* const* d_in, const int* in_sizes, int n_in,
                              void* d_out, int out_size, void* d_ws, size_t ws_size,
                              hipStream_t stream) {
  const int E = in_sizes[2] / 2;
  const int N = in_sizes[0] / 128;

  const float*     x   = (const float*)d_in[0];
  const float*     ea  = (const float*)d_in[1];
  const long long* ei  = (const long long*)d_in[2];
  const long long* row = ei;
  const long long* col = ei + E;
  const float* el_w1 = (const float*)d_in[3];  const float* el_b1 = (const float*)d_in[4];
  const float* el_w2 = (const float*)d_in[5];  const float* el_b2 = (const float*)d_in[6];
  const float* nl_w1 = (const float*)d_in[7];  const float* nl_b1 = (const float*)d_in[8];
  const float* nl_w2 = (const float*)d_in[9];  const float* nl_b2 = (const float*)d_in[10];
  const float* eu_w1 = (const float*)d_in[11]; const float* eu_b1 = (const float*)d_in[12];
  const float* eu_w2 = (const float*)d_in[13]; const float* eu_b2 = (const float*)d_in[14];
  const float* nu_w1 = (const float*)d_in[15]; const float* nu_b1 = (const float*)d_in[16];
  const float* nu_w2 = (const float*)d_in[17]; const float* nu_b2 = (const float*)d_in[18];
  const float* al_w1 = (const float*)d_in[19]; const float* al_b1 = (const float*)d_in[20];
  const float* al_w2 = (const float*)d_in[21]; const float* al_b2 = (const float*)d_in[22];
  const float* au_w1 = (const float*)d_in[23]; const float* au_b1 = (const float*)d_in[24];
  const float* au_w2 = (const float*)d_in[25]; const float* au_b2 = (const float*)d_in[26];

  char* p = (char*)d_ws;
  auto take = [&](size_t bytes) { char* r = p; p += (bytes + 255) & ~(size_t)255; return r; };
  float*    msg_l = (float*)   take((size_t)E * 129 * 4);
  float*    e_ij  = (float*)   take((size_t)E * 4);
  float*    ex_l  = (float*)   take((size_t)E * 4);
  float*    e_u   = (float*)   take((size_t)E * 4);
  float*    ex_u  = (float*)   take((size_t)E * 4);
  unsigned* max_l = (unsigned*)take((size_t)N * 4);
  float*    sum_l = (float*)   take((size_t)N * 4);
  unsigned* max_u = (unsigned*)take((size_t)N * 4);
  float*    sum_u = (float*)   take((size_t)N * 4);
  float*    agg_l = (float*)   take((size_t)N * 129 * 4);
  float*    agg_u = (float*)   take((size_t)N * 128 * 4);
  float*    nf_l  = (float*)   take((size_t)N * 128 * 4);
  __bf16* w1elf = (__bf16*)take(416 * 16 * 2);
  __bf16* w1alf = (__bf16*)take(416 * 16 * 2);
  __bf16* w1euf = (__bf16*)take(416 * 16 * 2);
  __bf16* w1auf = (__bf16*)take(416 * 16 * 2);
  __bf16* w1nlf = (__bf16*)take(288 * 16 * 2);
  __bf16* w1nuf = (__bf16*)take(256 * 16 * 2);
  __bf16* w2elf = (__bf16*)take(9 * 512 * 2);
  __bf16* w2euf = (__bf16*)take(8 * 512 * 2);
  __bf16* w2nlf = (__bf16*)take(8 * 512 * 2);
  __bf16* w2nuf = (__bf16*)take(8 * 512 * 2);

  // weight prep (fragment-swizzled bf16)
  prep_w1_frag<<<(416 * 16 + 255) / 256, 256, 0, stream>>>(el_w1, w1elf, 416, 1, 385);
  prep_w1_frag<<<(416 * 16 + 255) / 256, 256, 0, stream>>>(al_w1, w1alf, 416, 1, 385);
  prep_w1_frag<<<(416 * 16 + 255) / 256, 256, 0, stream>>>(eu_w1, w1euf, 416, 1, 385);
  prep_w1_frag<<<(416 * 16 + 255) / 256, 256, 0, stream>>>(au_w1, w1auf, 416, 1, 385);
  prep_w1_frag<<<(288 * 16 + 255) / 256, 256, 0, stream>>>(nl_w1, w1nlf, 288, 0, 257);
  prep_w1_frag<<<(256 * 16 + 255) / 256, 256, 0, stream>>>(nu_w1, w1nuf, 256, 0, 256);
  prep_w2_frag<<<(9 * 512 + 255) / 256, 256, 0, stream>>>(el_w2, w2elf, 129, 9);
  prep_w2_frag<<<(8 * 512 + 255) / 256, 256, 0, stream>>>(eu_w2, w2euf, 128, 8);
  prep_w2_frag<<<(8 * 512 + 255) / 256, 256, 0, stream>>>(nl_w2, w2nlf, 128, 8);
  prep_w2_frag<<<(8 * 512 + 255) / 256, 256, 0, stream>>>(nu_w2, w2nuf, 128, 8);

  auto zero = [&](void* q, long long n) {
    zero_kernel<<<(unsigned)((n + 255) / 256), 256, 0, stream>>>((float*)q, n);
  };
  zero(max_l, N); zero(sum_l, N); zero(max_u, N); zero(sum_u, N);
  zero(agg_l, (long long)N * 129); zero(agg_u, (long long)N * 128);

  const int etiles = (E + 15) / 16, eblocks = (etiles + 3) / 4;
  const int ntiles = (N + 15) / 16, nblocks = (ntiles + 3) / 4;
  const int egrid1d = (E + 255) / 256;

  // ---- lower pass ----
  edge_mlp_kernel<<<eblocks, 128, 0, stream>>>(ea, x, row, col,
      w1elf, el_b1, w2elf, el_b2, 129, 9,
      w1alf, al_b1, al_w2, al_b2, msg_l, e_ij, E);
  seg_max_kernel<<<egrid1d, 256, 0, stream>>>(e_ij, row, max_l, E);
  seg_expsum_kernel<<<egrid1d, 256, 0, stream>>>(e_ij, row, max_l, ex_l, sum_l, E);
  seg_agg_kernel<<<(E + 7) / 8, 256, 0, stream>>>(msg_l, ex_l, sum_l, row, agg_l, E, 129);
  node_mlp_kernel<<<nblocks, 128, 0, stream>>>(x, agg_l, 129,
      w1nlf, nl_b1, w2nlf, nl_b2, nf_l, N, 288);

  // ---- upper pass ----
  float* msg_u = (float*)d_out;
  float* nf_u  = (float*)d_out + (long long)E * 128;
  edge_mlp_kernel<<<eblocks, 128, 0, stream>>>(msg_l, nf_l, col, row,
      w1euf, eu_b1, w2euf, eu_b2, 128, 8,
      w1auf, au_b1, au_w2, au_b2, msg_u, e_u, E);
  seg_max_kernel<<<egrid1d, 256, 0, stream>>>(e_u, col, max_u, E);
  seg_expsum_kernel<<<egrid1d, 256, 0, stream>>>(e_u, col, max_u, ex_u, sum_u, E);
  seg_agg_kernel<<<(E + 7) / 8, 256, 0, stream>>>(msg_u, ex_u, sum_u, col, agg_u, E, 128);
  node_mlp_kernel<<<nblocks, 128, 0, stream>>>(nf_l, agg_u, 128,
      w1nuf, nu_b1, w2nuf, nu_b2, nf_u, N, 256);
}
